// UnifiedThinkingModule_64312840291131
// MI455X (gfx1250) — matrix operand order repaired and checked by reading, not compiled, not run on hardware
//
#include <hip/hip_runtime.h>
#include <hip/hip_bf16.h>
#include <math.h>
#include <stdint.h>

#define H     2048
#define NH    16
#define HD    128
#define NW    4
#define STEPS 5
#define NMEM  16384
#define EPS   1e-5f
#define SPAD  16400   // padded row count for K/V and scores stride

typedef __attribute__((ext_vector_type(16))) __bf16 v16bf;
typedef __attribute__((ext_vector_type(8)))  __bf16 v8bf;
typedef __attribute__((ext_vector_type(8)))  float  v8f;
typedef __attribute__((ext_vector_type(4)))  unsigned int v4u;
typedef __attribute__((ext_vector_type(8)))  int v8i;
typedef __attribute__((ext_vector_type(4)))  int v4i;

// ---------------------------------------------------------------- utilities
__device__ __forceinline__ float wave_sum(float v) {
#pragma unroll
  for (int m = 16; m >= 1; m >>= 1) v += __shfl_xor(v, m, 32);
  return v;
}

__global__ void copy_kernel(const float* __restrict__ src, float* __restrict__ dst, int n) {
  int i = blockIdx.x * 256 + threadIdx.x;
  if (i < n) dst[i] = src[i];
}

__global__ void zero_kernel(float* __restrict__ dst, int n) {
  int i = blockIdx.x * 256 + threadIdx.x;
  if (i < n) dst[i] = 0.0f;
}

// ------------------------------------------------- per-row layernorm stats
__global__ void row_stats_kernel(const float* __restrict__ x, float2* __restrict__ stats, int rows) {
  int wave = threadIdx.x >> 5, lane = threadIdx.x & 31;
  int row = blockIdx.x * 8 + wave;
  if (row >= rows) return;
  const float4* x4 = (const float4*)(x + (size_t)row * H);
  float s = 0.f, sq = 0.f;
  for (int i = lane; i < H / 4; i += 32) {
    float4 v = x4[i];
    s  += v.x + v.y + v.z + v.w;
    sq += v.x * v.x + v.y * v.y + v.z * v.z + v.w * v.w;
  }
  s = wave_sum(s); sq = wave_sum(sq);
  if (lane == 0) {
    float mu  = s / (float)H;
    float var = sq / (float)H - mu * mu;
    stats[row] = make_float2(mu, rsqrtf(var + EPS));
  }
}

// ------------------------------- LN(memory_states) -> bf16 A matrix (once)
__global__ void ln_to_bf16_kernel(const float* __restrict__ x, const float2* __restrict__ stats,
                                  const float* __restrict__ g, const float* __restrict__ b,
                                  __bf16* __restrict__ y) {
  size_t i = ((size_t)blockIdx.x * 256 + threadIdx.x) * 8;
  int row = (int)(i >> 11);
  int col = (int)(i & (H - 1));
  float2 st = stats[row];
  float4 v0 = *(const float4*)(x + i);
  float4 v1 = *(const float4*)(x + i + 4);
  float4 g0 = *(const float4*)(g + col);
  float4 g1 = *(const float4*)(g + col + 4);
  float4 b0 = *(const float4*)(b + col);
  float4 b1 = *(const float4*)(b + col + 4);
  y[i + 0] = (__bf16)((v0.x - st.x) * st.y * g0.x + b0.x);
  y[i + 1] = (__bf16)((v0.y - st.x) * st.y * g0.y + b0.y);
  y[i + 2] = (__bf16)((v0.z - st.x) * st.y * g0.z + b0.z);
  y[i + 3] = (__bf16)((v0.w - st.x) * st.y * g0.w + b0.w);
  y[i + 4] = (__bf16)((v1.x - st.x) * st.y * g1.x + b1.x);
  y[i + 5] = (__bf16)((v1.y - st.x) * st.y * g1.y + b1.y);
  y[i + 6] = (__bf16)((v1.z - st.x) * st.y * g1.z + b1.z);
  y[i + 7] = (__bf16)((v1.w - st.x) * st.y * g1.w + b1.w);
}

__global__ void f32_to_bf16_kernel(const float* __restrict__ x, __bf16* __restrict__ y, int n) {
  int i = (blockIdx.x * 256 + threadIdx.x) * 8;
  if (i >= n) return;
  float4 v0 = *(const float4*)(x + i);
  float4 v1 = *(const float4*)(x + i + 4);
  y[i + 0] = (__bf16)v0.x; y[i + 1] = (__bf16)v0.y;
  y[i + 2] = (__bf16)v0.z; y[i + 3] = (__bf16)v0.w;
  y[i + 4] = (__bf16)v1.x; y[i + 5] = (__bf16)v1.y;
  y[i + 6] = (__bf16)v1.z; y[i + 7] = (__bf16)v1.w;
}

// --------------------------------------------------------------------------
// Tensor Data Mover: async 2D tile load (128 rows x 64 bf16 cols) into LDS.
// D# per ISA cdna5 §8.3/8.4. LDS padding: +16B after every 128B stored
// (pad_interval code 4 = 32 DWORDs, pad_amount code 3 = 4 DWORDs) -> LDS row
// stride 144B (16B aligned, skews banks for conflict-free fragment loads).
// --------------------------------------------------------------------------
__device__ __forceinline__ void tdm_load_tile(const void* gptr, unsigned lds_off,
                                              unsigned tensor_rows) {
  unsigned long long ga = (unsigned long long)(uintptr_t)gptr;
  v4u g0;
  g0[0] = 1u;                                     // count=1, user descriptor
  g0[1] = lds_off;                                // lds_addr (bytes)
  g0[2] = (unsigned)(ga & 0xffffffffu);           // global_addr[31:0]
  g0[3] = (unsigned)((ga >> 32) & 0x01ffffffu) | 0x80000000u;  // addr[56:32] | type=2
  v8i g1;
  g1[0] = (int)((1u << 16) | (1u << 20) | (4u << 22) | (3u << 25)); // 2B elems, pad en
  g1[1] = (int)(((unsigned)H & 0xffffu) << 16);   // tensor_dim0[15:0] = 2048
  g1[2] = (int)((tensor_rows & 0xffffu) << 16);   // tensor_dim0 hi=0 | tensor_dim1 lo
  g1[3] = (int)((tensor_rows >> 16) | (64u << 16));   // tensor_dim1 hi | tile_dim0=64
  g1[4] = 128;                                    // tile_dim1 = 128 rows
  g1[5] = H;                                      // tensor_dim0_stride = 2048 elems
  g1[6] = 0;
  g1[7] = 0;
  v4i g2; v4i g3;
#pragma unroll
  for (int i = 0; i < 4; i++) { g2[i] = 0; g3[i] = 0; }
#if defined(__clang_major__) && __clang_major__ >= 23
  v8i g4;
#pragma unroll
  for (int i = 0; i < 8; i++) g4[i] = 0;
  __builtin_amdgcn_tensor_load_to_lds(g0, g1, g2, g3, g4, 0);
#else
  __builtin_amdgcn_tensor_load_to_lds(g0, g1, g2, g3, 0);
#endif
}

// --------------------------------------------------------------------------
// WMMA GEMM with double-buffered TDM pipeline: C[m,n] = sum_k A[m,k]*W[n,k]
// A,W bf16 row-major (inner dim contiguous = K). Block tile 128x128,
// 8 waves each doing 2x4 16x16 tiles; K staged 64/iter by the TDM.
// --------------------------------------------------------------------------
__global__ __launch_bounds__(256)
void gemm_tdm_wmma_kernel(const __bf16* __restrict__ A, const __bf16* __restrict__ W,
                          __bf16* __restrict__ C, int arows) {
  extern __shared__ char smem[];
  const unsigned TILE_BYTES = 128u * 144u;        // 128 rows * (128B data + 16B pad)

  const int tid  = threadIdx.x;
  const int wv   = tid >> 5;
  const int lane = tid & 31;
  const int wm   = (wv >> 1) * 32;   // 4 waves down M
  const int wn   = (wv & 1) * 64;    // 2 waves across N
  const int lr   = lane & 15;
  const bool hi  = lane >= 16;

  const int gm = blockIdx.y * 128;
  const int gn = blockIdx.x * 128;
  const __bf16* Abase = A + (size_t)gm * H;
  const __bf16* Wbase = W + (size_t)gn * H;

  v8f acc[2][4];
#pragma unroll
  for (int i = 0; i < 2; i++)
#pragma unroll
    for (int j = 0; j < 4; j++)
#pragma unroll
      for (int e = 0; e < 8; e++) acc[i][j][e] = 0.0f;

  if (wv == 0) {                                   // prologue: fill buffer 0
    tdm_load_tile(Abase, 0u, (unsigned)arows);
    tdm_load_tile(Wbase, 2u * TILE_BYTES, (unsigned)H);
  }

  const int nIter = H / 64;
  for (int it = 0; it < nIter; ++it) {
    const unsigned p = (unsigned)(it & 1);
    if (wv == 0) {
      if (it + 1 < nIter) {                        // issue next pair, then wait
        tdm_load_tile(Abase + (it + 1) * 64, (p ^ 1u) * TILE_BYTES, (unsigned)arows);
        tdm_load_tile(Wbase + (it + 1) * 64, 2u * TILE_BYTES + (p ^ 1u) * TILE_BYTES,
                      (unsigned)H);
        __builtin_amdgcn_s_wait_tensorcnt(2);      // current pair (oldest 2) done
      } else {
        __builtin_amdgcn_s_wait_tensorcnt(0);
      }
    }
    __syncthreads();                               // tile visible to all waves

    const char* Ab = smem + p * TILE_BYTES;
    const char* Bb = smem + 2u * TILE_BYTES + p * TILE_BYTES;
#pragma unroll
    for (int ks = 0; ks < 64; ks += 32) {
      // Preload ALL fragments (12 x ds_load_b128), then 8 back-to-back WMMAs.
      // A fragments: 16x32, ISA layout (lanes 0-15: K 0..7/16..23; 16-31: 8..15/24..31)
      v16bf af[2], bfr[4];
#pragma unroll
      for (int mt = 0; mt < 2; mt++) {
        const char* base = Ab + (wm + mt * 16 + lr) * 144 + (ks + (hi ? 8 : 0)) * 2;
        v8bf lo = *(const v8bf*)base;
        v8bf hv = *(const v8bf*)(base + 32);
#pragma unroll
        for (int i = 0; i < 8; i++) { af[mt][i] = lo[i]; af[mt][8 + i] = hv[i]; }
      }
#pragma unroll
      for (int nt = 0; nt < 4; nt++) {
        // B fragment 32x16: lane holds column n = lane&15, K 0..7/16..23 (lanes 0-15), 8..15/24..31 (16-31)
        const char* base = Bb + (wn + nt * 16 + lr) * 144 + (ks + (hi ? 8 : 0)) * 2;
        v8bf lo = *(const v8bf*)base;
        v8bf hv = *(const v8bf*)(base + 32);
#pragma unroll
        for (int i = 0; i < 8; i++) { bfr[nt][i] = lo[i]; bfr[nt][8 + i] = hv[i]; }
      }
#pragma unroll
      for (int nt = 0; nt < 4; nt++)
#pragma unroll
        for (int mt = 0; mt < 2; mt++)
          acc[mt][nt] = __builtin_amdgcn_wmma_f32_16x16x32_bf16(
              false, af[mt], false, bfr[nt], (short)0, acc[mt][nt], false, false);
    }
    __syncthreads();                               // done reading before overwrite
  }

  // store bf16 (C/D layout: VGPR r -> M = r + (lane>=16 ? 8 : 0), N = lane&15)
#pragma unroll
  for (int mt = 0; mt < 2; mt++)
#pragma unroll
    for (int nt = 0; nt < 4; nt++) {
      int n = gn + wn + nt * 16 + lr;
#pragma unroll
      for (int rr = 0; rr < 8; rr++) {
        int m = gm + wm + mt * 16 + rr + (hi ? 8 : 0);
        C[(size_t)m * H + n] = (__bf16)acc[mt][nt][rr];
      }
    }
}

// ----------------------------------- dynamic rows (working mem + thoughts)
__global__ void dyn_stats_kernel(const float* __restrict__ wm,
                                 const float* __restrict__ thoughts,
                                 float2* __restrict__ dynstats) {
  int row = blockIdx.x, lane = threadIdx.x;
  const float* src = (row < NW) ? (wm + (size_t)row * H) : (thoughts + (size_t)(row - NW) * H);
  const float4* x4 = (const float4*)src;
  float s = 0.f, sq = 0.f;
  for (int i = lane; i < H / 4; i += 32) {
    float4 v = x4[i];
    s  += v.x + v.y + v.z + v.w;
    sq += v.x * v.x + v.y * v.y + v.z * v.z + v.w * v.w;
  }
  s = wave_sum(s); sq = wave_sum(sq);
  if (lane == 0) {
    float mu = s / (float)H;
    dynstats[row] = make_float2(mu, rsqrtf(sq / (float)H - mu * mu + EPS));
  }
}

__global__ void dyn_kv_kernel(const float* __restrict__ wm, const float* __restrict__ thoughts,
                              const float2* __restrict__ dynstats,
                              const float* __restrict__ lng, const float* __restrict__ lnb,
                              const float* __restrict__ Wk, const float* __restrict__ Wv,
                              __bf16* __restrict__ Kmat, __bf16* __restrict__ Vmat) {
  int b   = blockIdx.x;
  int jc  = b & 255;           // 256 chunks of 8 output cols
  int mat = (b >> 8) & 1;
  int row = b >> 9;
  int wv_ = threadIdx.x >> 5, lane = threadIdx.x & 31;
  int j   = jc * 8 + wv_;
  const float* src = (row < NW) ? (wm + (size_t)row * H) : (thoughts + (size_t)(row - NW) * H);
  float2 st = dynstats[row];
  const float* Wm = mat ? Wv : Wk;
  const float4* x4 = (const float4*)src;
  const float4* g4 = (const float4*)lng;
  const float4* b4 = (const float4*)lnb;
  const float4* w4 = (const float4*)(Wm + (size_t)j * H);
  float acc = 0.f;
  for (int i = lane; i < H / 4; i += 32) {
    float4 x = x4[i], g = g4[i], bb = b4[i], w = w4[i];
    acc = fmaf((x.x - st.x) * st.y * g.x + bb.x, w.x, acc);
    acc = fmaf((x.y - st.x) * st.y * g.y + bb.y, w.y, acc);
    acc = fmaf((x.z - st.x) * st.y * g.z + bb.z, w.z, acc);
    acc = fmaf((x.w - st.x) * st.y * g.w + bb.w, w.w, acc);
  }
  acc = wave_sum(acc);
  if (lane == 0) {
    int srow = (row < NW) ? (NMEM + row) : (NMEM + NW + (row - NW));
    (mat ? Vmat : Kmat)[(size_t)srow * H + j] = (__bf16)acc;
  }
}

// ------------------------------------------------------ small vector kernels
__global__ void vec_ln_kernel(const float* __restrict__ x, const float* __restrict__ g,
                              const float* __restrict__ b, float* __restrict__ y) {
  __shared__ float red[256];
  int t = threadIdx.x;
  float s = 0.f, sq = 0.f;
  for (int i = t; i < H; i += 256) { float v = x[i]; s += v; sq += v * v; }
  red[t] = s; __syncthreads();
  for (int st = 128; st; st >>= 1) { if (t < st) red[t] += red[t + st]; __syncthreads(); }
  float mu = red[0] / (float)H; __syncthreads();
  red[t] = sq; __syncthreads();
  for (int st = 128; st; st >>= 1) { if (t < st) red[t] += red[t + st]; __syncthreads(); }
  float rstd = rsqrtf(red[0] / (float)H - mu * mu + EPS);
  for (int i = t; i < H; i += 256) y[i] = (x[i] - mu) * rstd * g[i] + b[i];
}

__global__ void add_ln_kernel(const float* __restrict__ a, const float* __restrict__ c,
                              const float* __restrict__ g, const float* __restrict__ b,
                              float* __restrict__ y1, float* __restrict__ y2) {
  __shared__ float red[256];
  int t = threadIdx.x;
  float s = 0.f, sq = 0.f;
  for (int i = t; i < H; i += 256) { float v = a[i] + c[i]; s += v; sq += v * v; }
  red[t] = s; __syncthreads();
  for (int st = 128; st; st >>= 1) { if (t < st) red[t] += red[t + st]; __syncthreads(); }
  float mu = red[0] / (float)H; __syncthreads();
  red[t] = sq; __syncthreads();
  for (int st = 128; st; st >>= 1) { if (t < st) red[t] += red[t + st]; __syncthreads(); }
  float rstd = rsqrtf(red[0] / (float)H - mu * mu + EPS);
  for (int i = t; i < H; i += 256) {
    float v = (a[i] + c[i] - mu) * rstd * g[i] + b[i];
    y1[i] = v; y2[i] = v;
  }
}

__global__ void matvec_kernel(const float* __restrict__ W, const float* __restrict__ x,
                              const float* __restrict__ bias, float* __restrict__ y, int R) {
  int wv_ = threadIdx.x >> 5, lane = threadIdx.x & 31;
  int row = blockIdx.x * 8 + wv_;
  if (row >= R) return;
  const float4* w4 = (const float4*)(W + (size_t)row * H);
  const float4* x4 = (const float4*)x;
  float acc = 0.f;
  for (int i = lane; i < H / 4; i += 32) {
    float4 w = w4[i], v = x4[i];
    acc = fmaf(w.x, v.x, acc); acc = fmaf(w.y, v.y, acc);
    acc = fmaf(w.z, v.z, acc); acc = fmaf(w.w, v.w, acc);
  }
  acc = wave_sum(acc);
  if (lane == 0) y[row] = acc + (bias ? bias[row] : 0.f);
}

// ------------------------------------------------------------- attention
__global__ void scores_kernel(const __bf16* __restrict__ K, const float* __restrict__ q,
                              float* __restrict__ scores, int S, float scale) {
  int wv_ = threadIdx.x >> 5, lane = threadIdx.x & 31;
  int s = blockIdx.x * 8 + wv_;
  if (s >= S) return;
  const v8bf* kr = (const v8bf*)(K + (size_t)s * H) + lane * 8;  // 64 contiguous elems/lane
  const float* qq = q + lane * 64;
  float acc = 0.f;
#pragma unroll
  for (int c = 0; c < 8; c++) {
    v8bf kv = kr[c];
#pragma unroll
    for (int i = 0; i < 8; i++) acc = fmaf((float)kv[i], qq[c * 8 + i], acc);
  }
  acc += __shfl_xor(acc, 1, 32);  // pair of lanes covers one 128-wide head
  if (!(lane & 1)) scores[(size_t)(lane >> 1) * SPAD + s] = acc * scale;
}

__global__ void softmax_kernel(float* __restrict__ scores, int S) {
  float* p = scores + (size_t)blockIdx.x * SPAD;
  __shared__ float red[256];
  int t = threadIdx.x;
  float m = -3.4e38f;
  for (int i = t; i < S; i += 256) m = fmaxf(m, p[i]);
  red[t] = m; __syncthreads();
  for (int st = 128; st; st >>= 1) { if (t < st) red[t] = fmaxf(red[t], red[t + st]); __syncthreads(); }
  m = red[0]; __syncthreads();
  float s = 0.f;
  for (int i = t; i < S; i += 256) s += expf(p[i] - m);
  red[t] = s; __syncthreads();
  for (int st = 128; st; st >>= 1) { if (t < st) red[t] += red[t + st]; __syncthreads(); }
  float inv = 1.f / red[0];
  for (int i = t; i < S; i += 256) p[i] = expf(p[i] - m) * inv;
}

__global__ void attnout_kernel(const __bf16* __restrict__ V, const float* __restrict__ attn,
                               float* __restrict__ out, int S) {
  int t  = threadIdx.x;
  int e0 = t * 8;         // 8 contiguous output elems per thread (2048 total)
  int hh = e0 >> 7;       // head index
  int s0 = blockIdx.x * 256;
  int s1 = min(s0 + 256, S);
  float acc[8];
#pragma unroll
  for (int i = 0; i < 8; i++) acc[i] = 0.f;
  const float* ap = attn + (size_t)hh * SPAD;
  for (int s = s0; s < s1; s++) {
    float a = ap[s];
    v8bf v = *(const v8bf*)(V + (size_t)s * H + e0);
#pragma unroll
    for (int i = 0; i < 8; i++) acc[i] = fmaf(a, (float)v[i], acc[i]);
  }
#pragma unroll
  for (int i = 0; i < 8; i++) atomicAdd(&out[e0 + i], acc[i]);
}

// --------------------------------------------------------------- GRU / gate
__global__ void gru_combine_kernel(const float* __restrict__ gi, const float* __restrict__ gh,
                                   float* __restrict__ q) {
  int i = blockIdx.x * 256 + threadIdx.x;
  if (i >= H) return;
  float r = 1.f / (1.f + expf(-(gi[i] + gh[i])));
  float z = 1.f / (1.f + expf(-(gi[H + i] + gh[H + i])));
  float n = tanhf(gi[2 * H + i] + r * gh[2 * H + i]);
  q[i] = (1.f - z) * n + z * q[i];
}

__global__ void gate_kernel(const float* __restrict__ Wg, const float* __restrict__ bg,
                            const float* __restrict__ thought, const float* __restrict__ q,
                            float* __restrict__ gate) {
  __shared__ float dots[4];
  int wv_ = threadIdx.x >> 5, lane = threadIdx.x & 31;
  const float* wr = Wg + (size_t)wv_ * 2 * H;
  float acc = 0.f;
  for (int i = lane; i < H; i += 32) acc = fmaf(wr[i], thought[i], acc);
  for (int i = lane; i < H; i += 32) acc = fmaf(wr[H + i], q[i], acc);
  acc = wave_sum(acc);
  if (!lane) dots[wv_] = acc + bg[wv_];
  __syncthreads();
  if (threadIdx.x == 0) {
    float m = fmaxf(fmaxf(dots[0], dots[1]), fmaxf(dots[2], dots[3]));
    float e0 = expf(dots[0] - m), e1 = expf(dots[1] - m);
    float e2 = expf(dots[2] - m), e3 = expf(dots[3] - m);
    float inv = 1.f / (e0 + e1 + e2 + e3);
    gate[0] = e0 * inv; gate[1] = e1 * inv; gate[2] = e2 * inv; gate[3] = e3 * inv;
  }
}

__global__ void wm_update_kernel(float* __restrict__ wm, const float* __restrict__ gate,
                                 const float* __restrict__ nc) {
  int i = blockIdx.x * 256 + threadIdx.x;   // 0 .. NW*H
  int r = i >> 11, hcol = i & (H - 1);
  float g = gate[r];
  wm[i] = wm[i] * (1.f - g) + g * nc[hcol];
}

// ---------------------------------------------------------------- launcher
extern "C" void kernel_launch(void* const* d_in, const int* in_sizes, int n_in,
                              void* d_out, int out_size, void* d_ws, size_t ws_size,
                              hipStream_t stream) {
  (void)in_sizes; (void)n_in; (void)out_size; (void)ws_size;
  const float* query_seed    = (const float*)d_in[0];
  const float* memory_states = (const float*)d_in[1];
  const float* Wq  = (const float*)d_in[2];
  const float* Wk  = (const float*)d_in[3];
  const float* Wv  = (const float*)d_in[4];
  const float* Wo  = (const float*)d_in[5];
  const float* gwih = (const float*)d_in[6];
  const float* gwhh = (const float*)d_in[7];
  const float* gbih = (const float*)d_in[8];
  const float* gbhh = (const float*)d_in[9];
  const float* Wg  = (const float*)d_in[10];
  const float* bg  = (const float*)d_in[11];
  const float* Wwork = (const float*)d_in[12];
  const float* lnqg = (const float*)d_in[13];
  const float* lnqb = (const float*)d_in[14];
  const float* lnkg = (const float*)d_in[15];
  const float* lnkb = (const float*)d_in[16];
  const float* lnog = (const float*)d_in[17];
  const float* lnob = (const float*)d_in[18];
  const float* init_work = (const float*)d_in[19];
  float* out = (float*)d_out;

  char* wsb = (char*)d_ws;
  size_t off = 0;
  auto take = [&](size_t bytes) -> void* {
    void* p = wsb + off;
    off += (bytes + 255) & ~(size_t)255;
    return p;
  };
  __bf16* Kmat   = (__bf16*)take((size_t)SPAD * H * 2);
  __bf16* Vmat   = (__bf16*)take((size_t)SPAD * H * 2);
  __bf16* Abf    = (__bf16*)take((size_t)NMEM * H * 2);   // LN'd memory, bf16
  __bf16* Wkbf   = (__bf16*)take((size_t)H * H * 2);
  __bf16* Wvbf   = (__bf16*)take((size_t)H * H * 2);
  float2* stats  = (float2*)take(sizeof(float2) * NMEM);
  float2* dynst  = (float2*)take(sizeof(float2) * 16);
  float*  scores = (float*)take(sizeof(float) * NH * SPAD);
  float*  qln    = (float*)take(H * 4);
  float*  qvec   = (float*)take(H * 4);
  float*  attno  = (float*)take(H * 4);
  float*  tpre   = (float*)take(H * 4);
  float*  thought= (float*)take(H * 4);
  float*  qbuf   = (float*)take(H * 4);
  float*  gi     = (float*)take(3 * H * 4);
  float*  gh     = (float*)take(3 * H * 4);
  float*  nc     = (float*)take(H * 4);
  float*  gate   = (float*)take(64);
  float*  wm     = (float*)take(NW * H * 4);

  // ---- one-time: K/V for the 16384 static memory rows (TDM + WMMA GEMMs)
  row_stats_kernel<<<NMEM / 8, 256, 0, stream>>>(memory_states, stats, NMEM);
  ln_to_bf16_kernel<<<NMEM * (H / 8) / 256, 256, 0, stream>>>(memory_states, stats,
                                                              lnkg, lnkb, Abf);
  f32_to_bf16_kernel<<<(H * H / 8) / 256, 256, 0, stream>>>(Wk, Wkbf, H * H);
  f32_to_bf16_kernel<<<(H * H / 8) / 256, 256, 0, stream>>>(Wv, Wvbf, H * H);
  dim3 gg(H / 128, NMEM / 128);
  const unsigned gemm_lds = 4u * 128u * 144u;   // 2 double-buffered 18KB tiles
  gemm_tdm_wmma_kernel<<<gg, 256, gemm_lds, stream>>>(Abf, Wkbf, Kmat, NMEM);
  gemm_tdm_wmma_kernel<<<gg, 256, gemm_lds, stream>>>(Abf, Wvbf, Vmat, NMEM);
  copy_kernel<<<(NW * H + 255) / 256, 256, 0, stream>>>(init_work, wm, NW * H);
  copy_kernel<<<(H + 255) / 256, 256, 0, stream>>>(query_seed, qbuf, H);

  const float scale = 1.0f / sqrtf((float)HD);
  float* thoughts_out = out + H;   // out = [query(H), thoughts(STEPS*H)]

  for (int t = 0; t < STEPS; t++) {
    int ndyn = NW + t;
    int S    = NMEM + NW + t;
    dyn_stats_kernel<<<ndyn, 32, 0, stream>>>(wm, thoughts_out, dynst);
    dyn_kv_kernel<<<ndyn * 512, 256, 0, stream>>>(wm, thoughts_out, dynst, lnkg, lnkb,
                                                  Wk, Wv, Kmat, Vmat);
    vec_ln_kernel<<<1, 256, 0, stream>>>(qbuf, lnqg, lnqb, qln);
    matvec_kernel<<<H / 8, 256, 0, stream>>>(Wq, qln, nullptr, qvec, H);
    scores_kernel<<<(S + 7) / 8, 256, 0, stream>>>(Kmat, qvec, scores, S, scale);
    softmax_kernel<<<NH, 256, 0, stream>>>(scores, S);
    zero_kernel<<<(H + 255) / 256, 256, 0, stream>>>(attno, H);
    attnout_kernel<<<(S + 255) / 256, 256, 0, stream>>>(Vmat, scores, attno, S);
    matvec_kernel<<<H / 8, 256, 0, stream>>>(Wo, attno, nullptr, tpre, H);
    add_ln_kernel<<<1, 256, 0, stream>>>(tpre, qbuf, lnog, lnob,
                                         thought, thoughts_out + (size_t)t * H);
    matvec_kernel<<<3 * H / 8, 256, 0, stream>>>(gwih, thought, gbih, gi, 3 * H);
    matvec_kernel<<<3 * H / 8, 256, 0, stream>>>(gwhh, qbuf, gbhh, gh, 3 * H);
    gru_combine_kernel<<<H / 256, 256, 0, stream>>>(gi, gh, qbuf);
    matvec_kernel<<<H / 8, 256, 0, stream>>>(Wwork, thought, nullptr, nc, H);
    gate_kernel<<<1, 128, 0, stream>>>(Wg, bg, thought, qbuf, gate);
    wm_update_kernel<<<NW * H / 256, 256, 0, stream>>>(wm, gate, nc);
  }
  copy_kernel<<<(H + 255) / 256, 256, 0, stream>>>(qbuf, out, H);
}
